// Net_10900626998029
// MI455X (gfx1250) — matrix operand order repaired and checked by reading, not compiled, not run on hardware
//
#include <hip/hip_runtime.h>
#include <hip/hip_bf16.h>

typedef __attribute__((ext_vector_type(16))) _Float16 v16h;
typedef __attribute__((ext_vector_type(8)))  float    v8f;

#define NB   128
#define NK   32
#define HWF  80
#define DWF  74
#define PIX  (HWF*HWF)
#define EPSL 1e-7f

__device__ __forceinline__ int iclamp(int v, int lo, int hi) {
    return v < lo ? lo : (v > hi ? hi : v);
}

// ---------------------------------------------------------------------------
// Weight repack: W [Cout][Cin*9] f32 -> per-lane A-fragment order, f16, padded.
// Fragment layout matches ISA 16-bit A 16x32: lane group g, VGPR v, half j:
//   kl = (v<4 ? 2v+j : 16+2(v-4)+j) + 8g
// Stored as Afrag[(mt*KS + ks)*32 + lane][16] so the conv does one v16h load.
// ---------------------------------------------------------------------------
__global__ __launch_bounds__(32)
void repack_weights(const float* __restrict__ W, _Float16* __restrict__ Afrag,
                    int Cout, int K, int KS)
{
    const int lane = threadIdx.x;
    const int z    = blockIdx.x;           // mt*KS + ks
    const int ks   = z % KS;
    const int mt   = z / KS;
    const int g    = lane >> 4;
    const int row  = mt * 16 + (lane & 15);
    v16h a;
#pragma unroll
    for (int h = 0; h < 16; ++h) {
        const int v  = h >> 1, j = h & 1;
        const int kl = (v < 4 ? (v * 2 + j) : (16 + (v - 4) * 2 + j)) + g * 8;
        const int k  = ks * 32 + kl;
        float wv = (row < Cout && k < K) ? W[row * K + k] : 0.0f;
        a[h] = (_Float16)wv;
    }
    *(v16h*)(Afrag + ((size_t)z * 32 + lane) * 16) = a;
}

// ---------------------------------------------------------------------------
// Implicit-GEMM 3x3 VALID conv via v_wmma_f32_16x16x32_f16.
// 160-thread block = 5 waves = the 5 x-tiles of one output row; one launch
// per Cout tile (MT is a template arg so the whole epilogue is branch-free
// with immediate row offsets). All im2col offsets fold to immediates; the
// half-wave select (g) is a single cndmask between two constants.
// ---------------------------------------------------------------------------
template<int Cin, int Cout, int Hin, int Win, int MT, bool RELU, bool INADD>
__global__ __launch_bounds__(160)
void conv3x3_wmma(const float* __restrict__ in, const _Float16* __restrict__ Afrag,
                  const float* __restrict__ bias, float* __restrict__ out)
{
    constexpr int Hout = Hin - 2, Wout = Win - 2;
    constexpr int K     = Cin * 9;
    constexpr int KS    = (K + 31) / 32;
    constexpr int cout0 = MT * 16;

    const int lane = threadIdx.x & 31;
    const int wave = threadIdx.x >> 5;             // x tile 0..4
    const int g    = lane >> 4;
    const int l15  = lane & 15;
    const int y    = blockIdx.y;
    const int b    = blockIdx.z;
    const int n    = wave * 16 + l15;              // output column
    const int nc   = n < Wout ? n : (Wout - 1);    // clamped for loads

    const float* __restrict__ inB =
        in + (size_t)b * Cin * Hin * Win + (size_t)y * Win + nc;

    v8f acc = {};
#pragma unroll
    for (int ks = 0; ks < KS; ++ks) {
        // A fragment: one contiguous 32-byte load from the repacked buffer.
        v16h a = *(const v16h*)(Afrag + (((size_t)MT * KS + ks) * 32 + lane) * 16);
        // B fragment: K = ks*32 + 8g + (h<8 ? h : h+8), N = lane&15. Offsets are immediates;
        // padded K selects offset 0 (annihilated by the zero weight).
        v16h bfr;
#pragma unroll
        for (int h = 0; h < 16; ++h) {
            const int k0   = ks * 32 + (h < 8 ? h : h + 8);  // g == 0
            const int k1   = k0 + 8;               // g == 1
            const int off0 = (k0 < K)
                ? ((k0 / 9) * Hin * Win + ((k0 % 9) / 3) * Win + (k0 % 3)) : 0;
            const int off1 = (k1 < K)
                ? ((k1 / 9) * Hin * Win + ((k1 % 9) / 3) * Win + (k1 % 3)) : 0;
            const int off  = g ? off1 : off0;
            float bv = inB[off];
            if (INADD) bv += -0.33f;
            bfr[h] = (_Float16)bv;
        }
        acc = __builtin_amdgcn_wmma_f32_16x16x32_f16(
                  false, a, false, bfr, (short)0, acc, false, false);
    }

    if (n < Wout) {
        float* __restrict__ outB =
            out + ((size_t)b * Cout * Hout + (size_t)y) * Wout + n;
#pragma unroll
        for (int r = 0; r < 8; ++r) {
            // D layout: VGPR r -> row r (lanes 0-15) / row r+8 (lanes 16-31).
            // Guards are compile-time for g known halves: emit only rows that
            // can exist; runtime part is just the g select.
            const int rowLo = cout0 + r;           // g == 0
            const int rowHi = cout0 + r + 8;       // g == 1
            const int row   = g ? rowHi : rowLo;
            const bool loOk = (rowLo < Cout);
            const bool hiOk = (rowHi < Cout);
            if (loOk && hiOk) {
                float v = acc[r] + bias[row];
                if (RELU) v = fmaxf(v, 0.0f);
                outB[(size_t)row * Hout * Wout] = v;
            } else if (loOk || hiOk) {
                if ((g == 0) == loOk) {            // only one half valid
                    float v = acc[r] + bias[row];
                    if (RELU) v = fmaxf(v, 0.0f);
                    outB[(size_t)row * Hout * Wout] = v;
                }
            }
        }
    }
}

// ---------------------------------------------------------------------------
// Block-wide reductions (256 threads).
// ---------------------------------------------------------------------------
__device__ float block_sum(float v, float* red) {
    const int tid = threadIdx.x;
    red[tid] = v; __syncthreads();
    for (int s = 128; s > 0; s >>= 1) {
        if (tid < s) red[tid] += red[tid + s];
        __syncthreads();
    }
    float r = red[0];
    __syncthreads();
    return r;
}
__device__ float block_max(float v, float* red) {
    const int tid = threadIdx.x;
    red[tid] = v; __syncthreads();
    for (int s = 128; s > 0; s >>= 1) {
        if (tid < s) red[tid] = fmaxf(red[tid], red[tid + s]);
        __syncthreads();
    }
    float r = red[0];
    __syncthreads();
    return r;
}

// ---------------------------------------------------------------------------
// Fused bilinear 74->80 upsample + T*softmax + soft-argmax.
// ---------------------------------------------------------------------------
__global__ __launch_bounds__(256)
void softargmax_kernel(const float* __restrict__ dense, float* __restrict__ kp,
                       float* __restrict__ mapOut, int writeMap)
{
    __shared__ float tile[PIX];
    __shared__ float red[256];
    const int bk  = blockIdx.x;
    const int tid = threadIdx.x;
    const float* __restrict__ d = dense + (size_t)bk * DWF * DWF;
    const float scale = (float)DWF / (float)HWF;

    for (int p = tid; p < PIX; p += 256) {
        const int py = p / HWF, px = p - py * HWF;
        const float sy = (py + 0.5f) * scale - 0.5f;
        const float sx = (px + 0.5f) * scale - 0.5f;
        const float fy = floorf(sy), fx = floorf(sx);
        const float wy = sy - fy,     wx = sx - fx;
        const int y0 = iclamp((int)fy,     0, DWF - 1);
        const int y1 = iclamp((int)fy + 1, 0, DWF - 1);
        const int x0 = iclamp((int)fx,     0, DWF - 1);
        const int x1 = iclamp((int)fx + 1, 0, DWF - 1);
        const float v00 = d[y0 * DWF + x0], v01 = d[y0 * DWF + x1];
        const float v10 = d[y1 * DWF + x0], v11 = d[y1 * DWF + x1];
        const float up = (1.f - wy) * ((1.f - wx) * v00 + wx * v01)
                       +        wy  * ((1.f - wx) * v10 + wx * v11);
        tile[p] = 400.0f * up;
    }
    __syncthreads();

    float m = -3.0e38f;
    for (int p = tid; p < PIX; p += 256) m = fmaxf(m, tile[p]);
    m = block_max(m, red);

    float s = 0.f, sc = 0.f, sr = 0.f;
    for (int p = tid; p < PIX; p += 256) {
        const int py = p / HWF, px = p - py * HWF;
        const float e = expf(tile[p] - m);
        s += e; sc += e * (float)px; sr += e * (float)py;
    }
    const float total = block_sum(s,  red);
    const float colW  = block_sum(sc, red);
    const float rowW  = block_sum(sr, red);
    if (tid == 0) {
        kp[bk * 2 + 0] = colW / total;   // x (col-weighted)
        kp[bk * 2 + 1] = rowW / total;   // y (row-weighted)
    }
    if (writeMap) {
        const float inv = 1.0f / total;
        for (int p = tid; p < PIX; p += 256)
            mapOut[(size_t)bk * PIX + p] = expf(tile[p] - m) * inv;
    }
}

// ---------------------------------------------------------------------------
// img_change = (sum_c |first_prev-first| > 0); also per-batch change sums.
// ---------------------------------------------------------------------------
__global__ __launch_bounds__(256)
void imgchange_kernel(const float* __restrict__ first, const float* __restrict__ firstp,
                      float* __restrict__ ic, float* __restrict__ changedSum)
{
    __shared__ float red[256];
    const int b = blockIdx.x, tid = threadIdx.x;
    float local = 0.f;
    for (int p = tid; p < PIX; p += 256) {
        float s = 0.f;
        for (int c = 0; c < 3; ++c) {
            const size_t idx = ((size_t)b * 3 + c) * PIX + p;
            s += fabsf(firstp[idx] - first[idx]);
        }
        const float v = (s > 0.f) ? 1.f : 0.f;
        ic[(size_t)b * PIX + p] = v;
        local += v;
    }
    const float t = block_sum(local, red);
    if (tid == 0) changedSum[b] = t;
}

__global__ __launch_bounds__(256)
void overlap_kernel(const float* __restrict__ map1, const float* __restrict__ ic,
                    float* __restrict__ overlap)
{
    __shared__ float red[256];
    const int bk = blockIdx.x, b = bk / NK, tid = threadIdx.x;
    float local = 0.f;
    for (int p = tid; p < PIX; p += 256)
        local += map1[(size_t)bk * PIX + p] * ic[(size_t)b * PIX + p];
    const float t = block_sum(local, red);
    if (tid == 0) overlap[bk] = t;
}

// ---------------------------------------------------------------------------
// Final losses: [kcl, scl, kvl, ssl]. Single 256-thread block.
// ---------------------------------------------------------------------------
__global__ __launch_bounds__(256)
void losses_kernel(const float* __restrict__ kp1, const float* __restrict__ kp1p,
                   const float* __restrict__ overlap, const float* __restrict__ changedSum,
                   float* __restrict__ out)
{
    __shared__ float red[256];
    const int tid = threadIdx.x;

    // keypoint consistency
    float a = 0.f;
    for (int i = tid; i < NB * NK; i += 256) {
        const float dx = kp1[2 * i]     - kp1p[2 * i];
        const float dy = kp1[2 * i + 1] - kp1p[2 * i + 1];
        a += dx * dx + dy * dy;
    }
    const float kcl = block_sum(a, red) / (float)(NB * NK);

    // silhouette losses
    float sclA = 0.f, sslA = 0.f, vA = 0.f;
    for (int b = tid; b < NB; b += 256) {
        const float valid = (changedSum[b] > 0.f) ? 1.f : 0.f;
        float sb = 0.f, sl = 0.f;
        for (int k = 0; k < NK; ++k) {
            const float ov = overlap[b * NK + k];
            sb += ov;
            sl += -logf(EPSL + ov);
        }
        sclA += valid * sl;
        sslA += valid * (-logf(EPSL + sb));
        vA   += valid;
    }
    float nv = block_sum(vA, red);
    nv = fmaxf(nv, 1.0f);
    const float scl = block_sum(sclA, red) / (nv * (float)NK);
    const float ssl = block_sum(sslA, red) / nv;

    // keypoint variety (faithful j<2 loop bug)
    float kv = 0.f;
    for (int t = tid; t < NB * NK * 2; t += 256) {
        const int j = t & 1;
        const int rest = t >> 1;
        const int i = rest % NK;
        const int b = rest / NK;
        if (i != j) {
            const float dx = kp1[(b * NK + i) * 2]     - kp1[(b * NK + j) * 2];
            const float dy = kp1[(b * NK + i) * 2 + 1] - kp1[(b * NK + j) * 2 + 1];
            const float d2 = dx * dx + dy * dy;
            kv += fmaxf(9.0f - d2, 0.0f);
        }
    }
    const float kvl = block_sum(kv, red) / ((float)NK * (float)NK * (float)NB);

    if (tid == 0) {
        out[0] = kcl; out[1] = scl; out[2] = kvl; out[3] = ssl;
    }
}

// ---------------------------------------------------------------------------
// Host-side pipeline.
// ---------------------------------------------------------------------------
static void run_encode(const float* x,
                       const _Float16* A1, const float* b1,
                       const _Float16* A2, const float* b2,
                       const _Float16* A3, const float* b3,
                       float* wsA, float* wsB, float* wsC,
                       float* kpOut, float* mapOut, int writeMap,
                       hipStream_t stream)
{
    // conv1: 3->20, 80->78, (x - 0.33), ReLU           (Cout tiles 0 and 1)
    conv3x3_wmma<3, 20, 80, 80, 0, true, true><<<dim3(1, 78, NB), 160, 0, stream>>>(
        x, A1, b1, wsA);
    conv3x3_wmma<3, 20, 80, 80, 1, true, true><<<dim3(1, 78, NB), 160, 0, stream>>>(
        x, A1, b1, wsA);
    // conv2: 20->30, 78->76, ReLU
    conv3x3_wmma<20, 30, 78, 78, 0, true, false><<<dim3(1, 76, NB), 160, 0, stream>>>(
        wsA, A2, b2, wsB);
    conv3x3_wmma<20, 30, 78, 78, 1, true, false><<<dim3(1, 76, NB), 160, 0, stream>>>(
        wsA, A2, b2, wsB);
    // conv3: 30->32, 76->74, no ReLU
    conv3x3_wmma<30, 32, 76, 76, 0, false, false><<<dim3(1, 74, NB), 160, 0, stream>>>(
        wsB, A3, b3, wsC);
    conv3x3_wmma<30, 32, 76, 76, 1, false, false><<<dim3(1, 74, NB), 160, 0, stream>>>(
        wsB, A3, b3, wsC);
    // upsample + softmax + soft-argmax
    softargmax_kernel<<<NB * NK, 256, 0, stream>>>(wsC, kpOut, mapOut, writeMap);
}

extern "C" void kernel_launch(void* const* d_in, const int* in_sizes, int n_in,
                              void* d_out, int out_size, void* d_ws, size_t ws_size,
                              hipStream_t stream)
{
    const float* first  = (const float*)d_in[0];
    const float* firstp = (const float*)d_in[1];
    const float* second = (const float*)d_in[2];
    const float* w1 = (const float*)d_in[3];
    const float* b1 = (const float*)d_in[4];
    const float* w2 = (const float*)d_in[5];
    const float* b2 = (const float*)d_in[6];
    const float* w3 = (const float*)d_in[7];
    const float* b3 = (const float*)d_in[8];

    float* out = (float*)d_out;
    // output layout (floats): kp1, kp2, kp1_prev, map1, map2, img_change, losses
    const size_t KP  = (size_t)NB * NK * 2;          // 8192
    const size_t MAP = (size_t)NB * NK * PIX;        // 26,214,400
    float* kp1     = out;
    float* kp2     = out + KP;
    float* kp1prev = out + 2 * KP;
    float* map1    = out + 3 * KP;
    float* map2    = map1 + MAP;
    float* imgch   = map2 + MAP;                     // [B,80,80]
    float* losses  = imgch + (size_t)NB * PIX;       // [4]

    // workspace layout (floats, then f16 fragment buffers)
    float* ws  = (float*)d_ws;
    float* wsA = ws;                                      // x2: 128*20*78*78
    float* wsB = wsA + (size_t)NB * 20 * 78 * 78;         // x3: 128*30*76*76
    float* wsC = wsB + (size_t)NB * 30 * 76 * 76;         // dense: 128*32*74*74
    float* changedSum = wsC + (size_t)NB * 32 * 74 * 74;  // [128]
    float* overlap    = changedSum + NB;                  // [4096]
    _Float16* A1 = (_Float16*)(overlap + NB * NK);        // 2*1*32*16 = 1024 h
    _Float16* A2 = A1 + 2 * 1 * 32 * 16;                  // 2*6*32*16 = 6144 h
    _Float16* A3 = A2 + 2 * 6 * 32 * 16;                  // 2*9*32*16 = 9216 h

    // repack weights into WMMA A-fragment order (f16, zero-padded)
    repack_weights<<<2 * 1, 32, 0, stream>>>(w1, A1, 20, 27,  1);
    repack_weights<<<2 * 6, 32, 0, stream>>>(w2, A2, 30, 180, 6);
    repack_weights<<<2 * 9, 32, 0, stream>>>(w3, A3, 32, 270, 9);

    // three encodes (first_prev's map is not an output -> writeMap=0)
    run_encode(first,  A1, b1, A2, b2, A3, b3, wsA, wsB, wsC, kp1,     map1, 1, stream);
    run_encode(firstp, A1, b1, A2, b2, A3, b3, wsA, wsB, wsC, kp1prev, map1, 0, stream);
    run_encode(second, A1, b1, A2, b2, A3, b3, wsA, wsB, wsC, kp2,     map2, 1, stream);

    imgchange_kernel<<<NB, 256, 0, stream>>>(first, firstp, imgch, changedSum);
    overlap_kernel<<<NB * NK, 256, 0, stream>>>(map1, imgch, overlap);
    losses_kernel<<<1, 256, 0, stream>>>(kp1, kp1prev, overlap, changedSum, losses);
}
